// BuddingLayer_84430467105102
// MI455X (gfx1250) — matrix operand order repaired and checked by reading, not compiled, not run on hardware
//
#include <hip/hip_runtime.h>
#include <hip/hip_bf16.h>

// ---------------------------------------------------------------------------
// BuddingLayer for MI455X (gfx1250, wave32):
//   out = (x with saturated cols zeroed) @ weight^T + bias
//         + sum_k relu( h2(x[:,k]) @ W3[k] + b3[k] )
// Dense path: bf16 WMMA (V_WMMA_F32_16X16X32_BF16), f32 accumulate.
// Bud path:   f32  WMMA (V_WMMA_F32_16X16X4_F32), rank-3 (padded to 4).
// ---------------------------------------------------------------------------

#define B_SZ 512
#define IN_SZ 2048
#define OUT_SZ 2048
#define K_BUDS 128

typedef __attribute__((ext_vector_type(16))) __bf16 v16bf;
typedef __attribute__((ext_vector_type(8)))  __bf16 v8bf;
typedef __attribute__((ext_vector_type(8)))  float  v8f;
typedef __attribute__((ext_vector_type(2)))  float  v2f;

// ---------------------------- prep kernels ---------------------------------

__global__ void k_mask_zero(int* __restrict__ mask) {
    int i = blockIdx.x * blockDim.x + threadIdx.x;
    if (i < IN_SZ) mask[i] = 0;
}

__global__ void k_mask_set(int* __restrict__ mask, const int* __restrict__ sat_idx) {
    int i = blockIdx.x * blockDim.x + threadIdx.x;
    if (i < K_BUDS) mask[sat_idx[i]] = 1;
}

// x (masked) -> bf16, row-major [B, IN]
__global__ void k_cvt_x(const float* __restrict__ x, const int* __restrict__ mask,
                        __bf16* __restrict__ xb) {
    int i = blockIdx.x * blockDim.x + threadIdx.x;
    int j = i & (IN_SZ - 1);
    float v = mask[j] ? 0.0f : x[i];
    xb[i] = (__bf16)v;
}

// weight -> bf16, row-major [OUT, IN]
__global__ void k_cvt_w(const float* __restrict__ w, __bf16* __restrict__ wb) {
    int i = blockIdx.x * blockDim.x + threadIdx.x;
    wb[i] = (__bf16)w[i];
}

// Per-(bud k, batch b): run the two 3x3 stages of the tiny MLP on scalar
// x[b, sat_idx[k]]. First layer input is x/3 broadcast to 3 equal lanes, so
// it collapses to v * rowsum(W1). Store h2 padded to 4 floats: [K, B, 4].
__global__ void k_h2(const float* __restrict__ x, const int* __restrict__ sat_idx,
                     const float* __restrict__ W1, const float* __restrict__ b1,
                     const float* __restrict__ W2, const float* __restrict__ b2,
                     float* __restrict__ H2) {
    int t = blockIdx.x * blockDim.x + threadIdx.x;
    int k = t >> 9;          // / B_SZ
    int b = t & (B_SZ - 1);
    float v = x[(size_t)b * IN_SZ + sat_idx[k]] * (1.0f / 3.0f);

    const float* W1k = W1 + k * 9;
    const float* b1k = b1 + k * 3;
    float h1[3];
#pragma unroll
    for (int j = 0; j < 3; ++j) {
        float s = W1k[j] + W1k[3 + j] + W1k[6 + j];
        h1[j] = fmaxf(v * s + b1k[j], 0.0f);
    }
    const float* W2k = W2 + k * 9;
    const float* b2k = b2 + k * 3;
    float* o = H2 + (size_t)(k * B_SZ + b) * 4;
#pragma unroll
    for (int j = 0; j < 3; ++j) {
        float s = h1[0] * W2k[j] + h1[1] * W2k[3 + j] + h1[2] * W2k[6 + j] + b2k[j];
        o[j] = fmaxf(s, 0.0f);
    }
    o[3] = 0.0f;
}

// ---------------------------- fused WMMA kernel ----------------------------
// One wave per 64x64 output tile. grid = (OUT/64, B/64), block = 32.
__global__ __launch_bounds__(32) void k_budding_gemm(
    const __bf16* __restrict__ xb,   // [B,  IN ] bf16 (masked)
    const __bf16* __restrict__ wb,   // [OUT,IN ] bf16
    const float*  __restrict__ bias, // [OUT]
    const float*  __restrict__ H2,   // [K, B, 4] f32 (padded)
    const float*  __restrict__ W3,   // [K, 3, OUT] f32
    const float*  __restrict__ b3,   // [K, OUT] f32
    float* __restrict__ out)         // [B, OUT] f32
{
    const int lane = threadIdx.x;
    const int rl = lane & 15;   // row-in-tile lane index (N for B/C/D, M for A)
    const int hf = lane >> 4;   // lane half select
    const int M0 = blockIdx.y * 64;
    const int N0 = blockIdx.x * 64;

    v8f acc[4][4];
    {
        v8f z = {0.f, 0.f, 0.f, 0.f, 0.f, 0.f, 0.f, 0.f};
#pragma unroll
        for (int mi = 0; mi < 4; ++mi)
#pragma unroll
            for (int ni = 0; ni < 4; ++ni) acc[mi][ni] = z;
    }

    // ---------------- dense path: bf16 WMMA, K loop over IN in steps of 32 --
    for (int kb = 0; kb < IN_SZ; kb += 32) {
        v16bf a[4], b[4];
#pragma unroll
        for (int mi = 0; mi < 4; ++mi) {
            // A 16x32 bf16 fragment: elems 0..15 = K kb+16*hf..kb+16*hf+15
            a[mi] = *(const v16bf*)(xb + (size_t)(M0 + mi * 16 + rl) * IN_SZ + kb + hf * 16);
        }
#pragma unroll
        for (int ni = 0; ni < 4; ++ni) {
            // B 32x16 bf16 fragment: lane rl = column n; 16 contiguous K values
            // starting at kb + 16*hf. B[k][n] = weight[N0+16*ni+n][k].
            b[ni] = *(const v16bf*)(wb + (size_t)(N0 + ni * 16 + rl) * IN_SZ + kb + hf * 16);
        }
#pragma unroll
        for (int mi = 0; mi < 4; ++mi)
#pragma unroll
            for (int ni = 0; ni < 4; ++ni)
                acc[mi][ni] = __builtin_amdgcn_wmma_f32_16x16x32_bf16(
                    false, a[mi], false, b[ni], (short)0, acc[mi][ni], false, false);
    }

    // ---------------- bud path: f32 WMMA 16x16x4, loop over 128 buds --------
    // Branch-free fragment build: keep EXEC uniform across the whole loop
    // (WMMA requires EXEC == all 1s; avoid exec-mask save/restore in the
    // inner loop). Upper-half lanes (hf==1) need K=3 == zero pad: load an
    // in-bounds dummy (w3k[0]) and multiply by 0 instead of branching.
    const float w1scale = (float)(1 - hf);          // 1 for lanes 0-15, 0 for 16-31
    const int   w1off   = hf ? 0 : OUT_SZ;          // address select, no branch
    for (int k = 0; k < K_BUDS; ++k) {
        v2f ab[4];
#pragma unroll
        for (int mi = 0; mi < 4; ++mi) {
            // A 16x4 f32 fragment: lanes 0-15 hold K=0,1; lanes 16-31 hold K=2,3.
            ab[mi] = *(const v2f*)(H2 + ((size_t)k * B_SZ + M0 + mi * 16 + rl) * 4 + 2 * hf);
        }
        v2f bb[4];
        float c0[4];
#pragma unroll
        for (int ni = 0; ni < 4; ++ni) {
            int o = N0 + ni * 16 + rl;
            const float* w3k = W3 + (size_t)k * 3 * OUT_SZ + o;
            // B 4x16 f32 fragment: lanes 0-15 hold K=0,1; lanes 16-31 hold K=2,3 (K=3 pad=0).
            float w0 = w3k[(2 * hf) * OUT_SZ];
            float w1 = w3k[w1off] * w1scale;
            v2f t = {w0, w1};
            bb[ni] = t;
            c0[ni] = b3[(size_t)k * OUT_SZ + o];
        }
#pragma unroll
        for (int mi = 0; mi < 4; ++mi)
#pragma unroll
            for (int ni = 0; ni < 4; ++ni) {
                float cv = c0[ni];
                v8f c = {cv, cv, cv, cv, cv, cv, cv, cv};
                v8f d = __builtin_amdgcn_wmma_f32_16x16x4_f32(
                    false, ab[mi], false, bb[ni], (short)0, c, false, false);
#pragma unroll
                for (int e = 0; e < 8; ++e)
                    acc[mi][ni][e] += fmaxf(d[e], 0.0f);
            }
    }

    // ---------------- epilogue: add dense bias, store ----------------------
#pragma unroll
    for (int ni = 0; ni < 4; ++ni) {
        int o = N0 + ni * 16 + rl;
        float bv = bias[o];
#pragma unroll
        for (int mi = 0; mi < 4; ++mi) {
#pragma unroll
            for (int e = 0; e < 8; ++e) {
                int m = M0 + mi * 16 + hf * 8 + e;
                out[(size_t)m * OUT_SZ + o] = acc[mi][ni][e] + bv;
            }
        }
    }
}

// ---------------------------------------------------------------------------

extern "C" void kernel_launch(void* const* d_in, const int* in_sizes, int n_in,
                              void* d_out, int out_size, void* d_ws, size_t ws_size,
                              hipStream_t stream) {
    (void)in_sizes; (void)n_in; (void)out_size; (void)ws_size;
    const float* x    = (const float*)d_in[0];
    const int*   sat  = (const int*)d_in[1];
    const float* wgt  = (const float*)d_in[2];
    const float* bias = (const float*)d_in[3];
    const float* W1   = (const float*)d_in[4];
    const float* b1   = (const float*)d_in[5];
    const float* W2   = (const float*)d_in[6];
    const float* b2   = (const float*)d_in[7];
    const float* W3   = (const float*)d_in[8];
    const float* b3   = (const float*)d_in[9];
    float* out = (float*)d_out;

    // workspace layout
    char* ws = (char*)d_ws;
    __bf16* wb   = (__bf16*)(ws);                                  // 8 MB
    __bf16* xb   = (__bf16*)(ws + (size_t)OUT_SZ * IN_SZ * 2);     // 2 MB
    float*  H2   = (float*)(ws + (size_t)OUT_SZ * IN_SZ * 2
                               + (size_t)B_SZ * IN_SZ * 2);        // 1 MB
    int*    mask = (int*)(ws + (size_t)OUT_SZ * IN_SZ * 2
                             + (size_t)B_SZ * IN_SZ * 2
                             + (size_t)K_BUDS * B_SZ * 4 * 4);     // 8 KB

    k_mask_zero<<<IN_SZ / 256, 256, 0, stream>>>(mask);
    k_mask_set<<<1, K_BUDS, 0, stream>>>(mask, sat);
    k_cvt_x<<<(B_SZ * IN_SZ) / 256, 256, 0, stream>>>(x, mask, xb);
    k_cvt_w<<<(OUT_SZ * IN_SZ) / 256, 256, 0, stream>>>(wgt, wb);
    k_h2<<<(K_BUDS * B_SZ) / 256, 256, 0, stream>>>(x, sat, W1, b1, W2, b2, H2);

    dim3 grid(OUT_SZ / 64, B_SZ / 64);  // (32, 8)
    k_budding_gemm<<<grid, 32, 0, stream>>>(xb, wb, bias, H2, W3, b3, out);
}
